// LatentNetwork_67104569033227
// MI455X (gfx1250) — matrix operand order repaired and checked by reading, not compiled, not run on hardware
//
#include <hip/hip_runtime.h>
#include <math.h>

// ---------------- problem constants (from reference setup_inputs) -----------
#define BB   32
#define NN   8192
#define LL   16
#define SS   128
#define TCC  256
#define HEADS 2
#define EPSF 1e-5f

// ---------------- bf16 / WMMA plumbing --------------------------------------
typedef __bf16 bf16_t;
typedef bf16_t v16bf __attribute__((ext_vector_type(16)));
typedef bf16_t v8bf  __attribute__((ext_vector_type(8)));
typedef float  v8f   __attribute__((ext_vector_type(8)));
typedef float  v4f   __attribute__((ext_vector_type(4)));

// pack two f32 -> packed bf16 pair (lo in [15:0]) with one v_perm_b32
__device__ __forceinline__ unsigned pack2bf(float lo, float hi) {
  union { float f; unsigned u; } a, b;
  a.f = lo; b.f = hi;
  return __builtin_amdgcn_perm(b.u + 0x8000u, a.u + 0x8000u, 0x07060302u);
}

// ---------------- WMMA GEMM: Y[b] = W (MxK) @ X[b] (KxN) ---------------------
// block = 256 threads = 8 waves (2x4); block tile 64x128; each wave owns a
// 32x32 patch = 2x2 WMMA 16x16 tiles; K step 32 (bf16 WMMA depth).
#define KSTEP 32
#define MT    64
#define NT    128
#define STRA  40   // LDS row stride (elements) for A [64][40]  (16B-aligned rows)
#define STRB  40   // LDS row stride for B transposed [128][40]

template<bool GUARD>
__global__ __launch_bounds__(256)
void k_gemm_wmma(const float* __restrict__ W, const float* __restrict__ X,
                 float* __restrict__ Y, int M, int K, int N)
{
  __shared__ unsigned short lA[MT * STRA];
  __shared__ unsigned short lB[NT * STRB];
  const int b  = blockIdx.z;
  const int m0 = blockIdx.y * MT;
  const int n0 = blockIdx.x * NT;
  const float* __restrict__ Xb = X + (size_t)b * K * N;
  float* __restrict__ Yb       = Y + (size_t)b * M * N;
  const int tid  = threadIdx.x;
  const int wave = tid >> 5;
  const int lane = tid & 31;
  const int wm = wave >> 2, wn = wave & 3;      // 2 x 4 waves of 32x32 patches
  const int half = lane >> 4, lr = lane & 15;
  v8f acc[2][2] = {};

  for (int k0 = 0; k0 < K; k0 += KSTEP) {
    { // ---- stage A tile (64m x 32k) as bf16 [m][k]; 8 elems/thread ----------
      const int idx = tid << 3;                 // 2048 elements
      const int mi = idx >> 5, kk = idx & 31;   // kk in {0,8,16,24}
      unsigned int* dst = (unsigned int*)&lA[mi * STRA + kk];
      if (!GUARD) {
        const float* wr = W + (size_t)(m0 + mi) * K + k0 + kk;
        v4f f0 = *(const v4f*)wr;
        v4f f1 = *(const v4f*)(wr + 4);
        dst[0] = pack2bf(f0[0], f0[1]);
        dst[1] = pack2bf(f0[2], f0[3]);
        dst[2] = pack2bf(f1[0], f1[1]);
        dst[3] = pack2bf(f1[2], f1[3]);
      } else {
        const int gm = m0 + mi;
        const int gmc = gm < M ? gm : M - 1;    // clamp; select 0 after load
        float fv[8];
#pragma unroll
        for (int j = 0; j < 8; ++j) {
          const int gk  = k0 + kk + j;
          const int gkc = gk < K ? gk : K - 1;
          const float v = W[(size_t)gmc * K + gkc];
          fv[j] = (gm < M && gk < K) ? v : 0.f;
        }
#pragma unroll
        for (int j = 0; j < 4; ++j)
          dst[j] = pack2bf(fv[2 * j], fv[2 * j + 1]);
      }
    }
    // ---- stage B tile (32k x 128n) transposed [n][k], packed k-pairs --------
    if (!GUARD) {
#pragma unroll
      for (int r = 0; r < 2; ++r) {             // 512 items of 4 packed u32
        const int item = tid + (r << 8);
        const int p  = item >> 5;               // k-pair 0..15
        const int nq = (item & 31) << 2;        // n quad base
        const float* xr = Xb + (size_t)(k0 + 2 * p) * N + n0 + nq;
        v4f f0 = *(const v4f*)xr;               // row k = 2p
        v4f f1 = *(const v4f*)(xr + N);         // row k = 2p+1
#pragma unroll
        for (int j = 0; j < 4; ++j)
          *(unsigned int*)&lB[(nq + j) * STRB + 2 * p] = pack2bf(f0[j], f1[j]);
      }
    } else {
#pragma unroll
      for (int r = 0; r < 8; ++r) {             // 2048 u32 / 256 threads
        const int idx = tid + (r << 8);
        const int p = idx >> 7;                 // k-pair 0..15
        const int n = idx & 127;
        const int gk = k0 + 2 * p, gn = n0 + n;
        const int gnc  = gn < N ? gn : N - 1;
        const int gk0c = gk < K ? gk : K - 1;
        const int gk1c = gk + 1 < K ? gk + 1 : K - 1;
        float f0 = Xb[(size_t)gk0c * N + gnc];
        float f1 = Xb[(size_t)gk1c * N + gnc];
        if (!(gk < K && gn < N))     f0 = 0.f;
        if (!(gk + 1 < K && gn < N)) f1 = 0.f;
        *(unsigned int*)&lB[n * STRB + 2 * p] = pack2bf(f0, f1);
      }
    }
    __syncthreads();

    if (!GUARD && k0 + KSTEP < K) {             // prefetch next X tile
      const int pk2 = k0 + KSTEP + (tid >> 6);
      const int pn  = n0 + ((tid & 63) << 1);
      __builtin_prefetch(&Xb[(size_t)pk2 * N + pn], 0, 1);
    }

    // ---- fragments: A 16x32 (two b128 per frag), B 32x16 (two b128) --------
    v16bf av[2], bv[2];
#pragma unroll
    for (int t = 0; t < 2; ++t) {
      const unsigned short* pa = &lA[(wm * 32 + t * 16 + lr) * STRA + 8 * half];
      v8bf a0 = *(const v8bf*)(pa);
      v8bf a1 = *(const v8bf*)(pa + 16);
      av[t] = __builtin_shufflevector(a0, a1, 0,1,2,3,4,5,6,7,8,9,10,11,12,13,14,15);
      const unsigned short* pb = &lB[(wn * 32 + t * 16 + lr) * STRB + 8 * half];
      v8bf b0 = *(const v8bf*)(pb);
      v8bf b1 = *(const v8bf*)(pb + 16);
      bv[t] = __builtin_shufflevector(b0, b1, 0,1,2,3,4,5,6,7,8,9,10,11,12,13,14,15);
    }
#pragma unroll
    for (int i = 0; i < 2; ++i)
#pragma unroll
      for (int j = 0; j < 2; ++j)
        acc[i][j] = __builtin_amdgcn_wmma_f32_16x16x32_bf16(false, av[i], false, bv[j],
                                                            (short)0, acc[i][j],
                                                            false, false);
    __syncthreads();
  }

#pragma unroll
  for (int i = 0; i < 2; ++i)
#pragma unroll
    for (int j = 0; j < 2; ++j)
#pragma unroll
      for (int vv = 0; vv < 8; ++vv) {          // C layout: VGPR v -> row v + 8*half
        const int row = m0 + wm * 32 + i * 16 + half * 8 + vv;
        const int col = n0 + wn * 32 + j * 16 + lr;
        if (!GUARD || (row < M && col < N))
          Yb[(size_t)row * N + col] = acc[i][j][vv];
      }
}

// ---------------- instance norm: stats + apply -------------------------------
__global__ __launch_bounds__(256)
void k_stats(const float* __restrict__ X, float* __restrict__ mean,
             float* __restrict__ rstd, int N)
{
  const size_t row = blockIdx.x;
  const float* p = X + row * (size_t)N;
  float s = 0.f, s2 = 0.f;
  for (int i = threadIdx.x; i < N; i += 256) { float v = p[i]; s += v; s2 += v * v; }
  __shared__ float sh0[256], sh1[256];
  sh0[threadIdx.x] = s; sh1[threadIdx.x] = s2;
  __syncthreads();
  for (int off = 128; off > 0; off >>= 1) {
    if (threadIdx.x < off) { sh0[threadIdx.x] += sh0[threadIdx.x + off];
                             sh1[threadIdx.x] += sh1[threadIdx.x + off]; }
    __syncthreads();
  }
  if (threadIdx.x == 0) {
    float m = sh0[0] / (float)N;
    float v = sh1[0] / (float)N - m * m;
    mean[row] = m;
    rstd[row] = rsqrtf(v + EPSF);
  }
}

__global__ void k_norm_apply(const float* __restrict__ X, const float* __restrict__ mean,
                             const float* __restrict__ rstd, const float* res,
                             float* out, int N, int relu)
{
  const int row = blockIdx.y;
  const int i = blockIdx.x * 256 + threadIdx.x;
  if (i >= N) return;
  const size_t idx = (size_t)row * N + i;
  float v = (X[idx] - mean[row]) * rstd[row];
  if (relu) v = fmaxf(v, 0.f);
  if (res)  v += res[idx];
  out[idx] = v;
}

// ---------------- KNN gather + max over S ------------------------------------
__global__ void k_gather_max(const float* __restrict__ feat, const int* __restrict__ knn,
                             float* __restrict__ tok, int C, int N, int L, int S)
{
  const int b = blockIdx.z, l = blockIdx.y;
  const int c = blockIdx.x * 128 + threadIdx.x;
  if (c >= C) return;
  const int* kn = knn + ((size_t)b * L + l) * S;
  const float* f = feat + ((size_t)b * C + c) * N;
  float m = -3.4e38f;
  for (int s = 0; s < S; ++s) m = fmaxf(m, f[kn[s]]);
  tok[((size_t)b * C + c) * L + l] = m;
}

// ---------------- token transformer attention (B x 2 heads, L=16) ------------
__global__ __launch_bounds__(256)
void k_tok_attn(const float* __restrict__ kb_, const float* __restrict__ qb_,
                const float* __restrict__ vb_, float* __restrict__ out)
{
  const int b = blockIdx.x, h = blockIdx.y;
  const int CK = TCC / 2 / HEADS;   // 64
  const int CV = TCC / HEADS;       // 128
  __shared__ float sk[64 * 16], sq[64 * 16], sv[128 * 16], skq[16 * 16];
  const int tid = threadIdx.x;
  const float* kb = kb_ + ((size_t)b * (TCC / 2) + h * CK) * LL;
  const float* qb = qb_ + ((size_t)b * (TCC / 2) + h * CK) * LL;
  const float* vb = vb_ + ((size_t)b * TCC + h * CV) * LL;
  for (int i = tid; i < CK * LL; i += 256) { sk[i] = kb[i]; sq[i] = qb[i]; }
  for (int i = tid; i < CV * LL; i += 256) sv[i] = vb[i];
  __syncthreads();
  { // kq[l][m] = sum_c k[c][l] q[c][m] / sqrt(64)
    int l = tid >> 4, m = tid & 15;
    float s = 0.f;
#pragma unroll 4
    for (int c = 0; c < CK; ++c) s += sk[c * LL + l] * sq[c * LL + m];
    skq[l * 16 + m] = s * 0.125f;
  }
  __syncthreads();
  if (tid < 16) {                   // softmax over l per column m
    int m = tid;
    float mx = -1e30f;
    for (int l = 0; l < 16; ++l) mx = fmaxf(mx, skq[l * 16 + m]);
    float ss = 0.f;
    for (int l = 0; l < 16; ++l) { float e = __expf(skq[l * 16 + m] - mx); skq[l * 16 + m] = e; ss += e; }
    float inv = 1.f / ss;
    for (int l = 0; l < 16; ++l) skq[l * 16 + m] *= inv;
  }
  __syncthreads();
  float* ob = out + ((size_t)b * TCC + h * CV) * LL;
#pragma unroll
  for (int r = 0; r < 8; ++r) {     // kqv[c][m] = sum_l v[c][l] kq[l][m]
    int idx = tid + (r << 8);
    int c = idx >> 4, m = idx & 15;
    float s = 0.f;
#pragma unroll
    for (int l = 0; l < 16; ++l) s += sv[c * LL + l] * skq[l * 16 + m];
    ob[(size_t)c * LL + m] = s;
  }
}

// ---------------- projector attention over N points --------------------------
__global__ __launch_bounds__(256)
void k_proj_attn(const float* __restrict__ q, const float* __restrict__ k,
                 const float* __restrict__ v, float* __restrict__ out,
                 int C, int N, int L)
{
  __shared__ float sk[64 * 16], sv[64 * 16];
  const int b = blockIdx.z, h = blockIdx.y;
  const int ch = C >> 1;            // per-head channels (<=64)
  const int tid = threadIdx.x;
  const float* kb = k + ((size_t)b * C + h * ch) * L;
  const float* vb = v + ((size_t)b * C + h * ch) * L;
  for (int i = tid; i < ch * L; i += 256) { sk[i] = kb[i]; sv[i] = vb[i]; }
  __syncthreads();
  const int s = blockIdx.x * 256 + tid;
  if (s >= N) return;
  const float* qb = q + ((size_t)b * C + h * ch) * N + s;
  float dots[16];
#pragma unroll
  for (int l = 0; l < 16; ++l) dots[l] = 0.f;
  for (int c = 0; c < ch; ++c) {
    float qv = qb[(size_t)c * N];
#pragma unroll
    for (int l = 0; l < 16; ++l) dots[l] += qv * sk[c * 16 + l];
  }
  const float scale = rsqrtf((float)ch);
  float mx = -1e30f;
#pragma unroll
  for (int l = 0; l < 16; ++l) { dots[l] *= scale; mx = fmaxf(mx, dots[l]); }
  float ss = 0.f;
#pragma unroll
  for (int l = 0; l < 16; ++l) { dots[l] = __expf(dots[l] - mx); ss += dots[l]; }
  const float inv = 1.f / ss;
#pragma unroll
  for (int l = 0; l < 16; ++l) dots[l] *= inv;
  float* ob = out + ((size_t)b * C + h * ch) * N + s;
  for (int c = 0; c < ch; ++c) {
    float acc = 0.f;
#pragma unroll
    for (int l = 0; l < 16; ++l) acc += sv[c * 16 + l] * dots[l];
    ob[(size_t)c * N] = acc;
  }
}

// ---------------- decoder concat builder -------------------------------------
// out (B, 3+256+48+16 = 323, L*S): [coords-gather | tok bcast | kp bcast | eye(L)]
__global__ void k_build_dec(const float* __restrict__ x, const int* __restrict__ knn,
                            const float* __restrict__ tokens, const float* __restrict__ kp,
                            float* __restrict__ out, int N, int L, int S)
{
  const int b = blockIdx.z;
  const int ls = blockIdx.x * 256 + threadIdx.x;
  if (ls >= L * S) return;
  const int l = ls / S, s = ls % S;
  const int LS = L * S;
  const int Ctot = 3 + TCC + 3 * LL + LL;     // 323
  float* o = out + (size_t)b * Ctot * LS;
  const int g = knn[((size_t)b * L + l) * S + s];
  for (int c = 0; c < 3; ++c)
    o[(size_t)c * LS + ls] = x[((size_t)b * 3 + c) * N + g];
  for (int c = 0; c < TCC; ++c)
    o[(size_t)(3 + c) * LS + ls] = tokens[((size_t)b * TCC + c) * L + l];
  for (int c = 0; c < 3 * LL; ++c)
    o[(size_t)(3 + TCC + c) * LS + ls] = kp[(size_t)b * 3 * LL + c];
  for (int c = 0; c < LL; ++c)
    o[(size_t)(3 + TCC + 3 * LL + c) * LS + ls] = (c == l) ? 1.f : 0.f;
}

// ---------------- host orchestration -----------------------------------------
struct Rim {
  const float *d1, *d2, *fb, *k, *pf1, *pf2, *pk, *pq, *pv, *q, *tf1, *tf2, *v;
};

static Rim get_rim(void* const* din, int base, bool has_fb) {
  Rim r; int i = base;
  r.d1 = (const float*)din[i++]; r.d2 = (const float*)din[i++];
  r.fb = has_fb ? (const float*)din[i++] : nullptr;
  r.k  = (const float*)din[i++];
  r.pf1 = (const float*)din[i++]; r.pf2 = (const float*)din[i++];
  r.pk = (const float*)din[i++]; r.pq = (const float*)din[i++];
  r.pv = (const float*)din[i++]; r.q  = (const float*)din[i++];
  r.tf1 = (const float*)din[i++]; r.tf2 = (const float*)din[i++];
  r.v  = (const float*)din[i++];
  return r;
}

extern "C" void kernel_launch(void* const* d_in, const int* in_sizes, int n_in,
                              void* d_out, int out_size, void* d_ws, size_t ws_size,
                              hipStream_t stream)
{
  (void)in_sizes; (void)n_in; (void)out_size; (void)ws_size;

  // inputs in jax-tree (sorted-key) order
  const float* xin  = (const float*)d_in[0];       // (B,3,N)
  const float* kp   = (const float*)d_in[1];       // (B,3,L)
  const int*   knn  = (const int*)d_in[2];         // (B,L,S)
  const float* dec1 = (const float*)d_in[3];       // (128,323)
  const float* dec2 = (const float*)d_in[4];       // (64,128)
  const float* dec3 = (const float*)d_in[5];       // (32,64)
  const float* dec4 = (const float*)d_in[6];       // (3,32)
  Rim s1a = get_rim(d_in, 7,  false);
  Rim s1b = get_rim(d_in, 19, true);
  Rim s2a = get_rim(d_in, 32, false);
  Rim s2b = get_rim(d_in, 44, true);
  Rim s3a = get_rim(d_in, 57, false);
  Rim s3b = get_rim(d_in, 69, false);
  const float* stem1 = (const float*)d_in[81];     // (32,3)
  const float* stem2 = (const float*)d_in[82];     // (32,32)

  // workspace allocator
  float* ws = (float*)d_ws;
  size_t off = 0;
  auto alloc = [&](size_t n) { float* p = ws + off; off += n; return p; };
  float* f    = alloc((size_t)BB * 128 * NN);   // feature map (<=128 ch)
  float* g0   = alloc((size_t)BB * 256 * NN);   // big temp (<=256 ch)
  float* g1   = alloc((size_t)BB * 128 * NN);
  float* g2   = alloc((size_t)BB * 128 * NN);
  float* tokens = alloc((size_t)BB * TCC * LL);
  float* tk   = alloc((size_t)BB * 128 * LL);
  float* tq   = alloc((size_t)BB * 128 * LL);
  float* tv   = alloc((size_t)BB * 256 * LL);
  float* th   = alloc((size_t)BB * 512 * LL);
  float* tt   = alloc((size_t)BB * 256 * LL);
  float* tkqv = alloc((size_t)BB * 256 * LL);
  float* mean = alloc((size_t)BB * 512);
  float* rstd = alloc((size_t)BB * 512);

  auto gemm = [&](const float* W, const float* X, float* Y, int M, int K, int N) {
    dim3 g((N + NT - 1) / NT, (M + MT - 1) / MT, BB);
    const bool fast = (M % MT == 0) && (N % NT == 0) && (K % KSTEP == 0);
    if (fast) k_gemm_wmma<false><<<g, 256, 0, stream>>>(W, X, Y, M, K, N);
    else      k_gemm_wmma<true ><<<g, 256, 0, stream>>>(W, X, Y, M, K, N);
  };
  auto norm = [&](float* X, int rows, int N, const float* res, float* out, int relu) {
    k_stats<<<rows, 256, 0, stream>>>(X, mean, rstd, N);
    dim3 g((N + 255) / 256, rows);
    k_norm_apply<<<g, 256, 0, stream>>>(X, mean, rstd, res, out, N, relu);
  };
  auto conv_in = [&](const float* W, const float* X, float* out,
                     int M, int K, int N, int relu, const float* res) {
    gemm(W, X, out, M, K, N);
    norm(out, BB * M, N, res, out, relu);
  };

  auto run_rim = [&](const Rim& p, int inc, int outc, bool first) {
    // tok = max_S gather(f)  -> tt (B,inc,L)
    dim3 gg((inc + 127) / 128, LL, BB);
    k_gather_max<<<gg, 128, 0, stream>>>(f, knn, tt, inc, NN, LL, SS);
    // tok = in1(d2 @ relu(in1(d1 @ tok))) (+ prev tokens)
    conv_in(p.d1, tt, th, TCC, inc, LL, 1, nullptr);
    gemm(p.d2, th, tt, TCC, TCC, LL);
    norm(tt, BB * TCC, LL, first ? nullptr : tokens, tokens, 0);
    // transformer on tokens
    conv_in(p.k, tokens, tk, TCC / 2, TCC, LL, 0, nullptr);
    conv_in(p.q, tokens, tq, TCC / 2, TCC, LL, 0, nullptr);
    conv_in(p.v, tokens, tv, TCC,     TCC, LL, 0, nullptr);
    k_tok_attn<<<dim3(BB, HEADS), 256, 0, stream>>>(tk, tq, tv, tkqv);
    norm(tkqv, BB * TCC, LL, tokens, tokens, 0);           // x = x + in1(kqv)
    conv_in(p.tf1, tokens, th, 2 * TCC, TCC, LL, 1, nullptr);
    gemm(p.tf2, th, tt, TCC, 2 * TCC, LL);
    norm(tt, BB * TCC, LL, tokens, tokens, 0);             // x = x + ff1d(x)
    // projector on feat
    const float* xf = f;
    if (p.fb) { conv_in(p.fb, f, g1, outc, inc, NN, 0, nullptr); xf = g1; }
    conv_in(p.pv, tokens, tv, outc, TCC, LL, 0, nullptr);
    conv_in(p.pk, tokens, tk, outc, TCC, LL, 0, nullptr);
    conv_in(p.pq, xf, g2, outc, outc, NN, 0, nullptr);
    dim3 ga((NN + 255) / 256, HEADS, BB);
    k_proj_attn<<<ga, 256, 0, stream>>>(g2, tk, tv, g0, outc, NN, LL);
    norm(g0, BB * outc, NN, xf, g2, 0);                    // t = x + in1(x_p)
    conv_in(p.pf1, g2, g0, 2 * outc, outc, NN, 1, nullptr);
    gemm(p.pf2, g0, g2, outc, 2 * outc, NN);
    norm(g2, BB * outc, NN, xf, f, 0);                     // f = x + in1(...)
  };

  // stem
  conv_in(stem1, xin, g1, 32, 3, NN, 1, nullptr);
  conv_in(stem2, g1, f, 32, 32, NN, 0, nullptr);

  run_rim(s1a, 32, 32, true);
  run_rim(s1b, 32, 64, false);
  run_rim(s2a, 64, 64, false);
  run_rim(s2b, 64, 128, false);
  run_rim(s3a, 128, 128, false);
  run_rim(s3b, 128, 128, false);

  // decoder
  const int LS = LL * SS;                                   // 2048
  const int Cdec = 3 + TCC + 3 * LL + LL;                   // 323
  k_build_dec<<<dim3((LS + 255) / 256, 1, BB), 256, 0, stream>>>(xin, knn, tokens, kp, g0, NN, LL, SS);
  conv_in(dec1, g0, g1, 128, Cdec, LS, 1, nullptr);
  conv_in(dec2, g1, g2, 64, 128, LS, 1, nullptr);
  conv_in(dec3, g2, g1, 32, 64, LS, 1, nullptr);
  gemm(dec4, g1, (float*)d_out, 3, 32, LS);                 // recon straight to d_out

  // outputs: (recon, f, tokens) concatenated flat
  float* outp = (float*)d_out;
  const size_t recon_n = (size_t)BB * 3 * LS;
  hipMemcpyAsync(outp + recon_n, f, (size_t)BB * 128 * NN * sizeof(float),
                 hipMemcpyDeviceToDevice, stream);
  hipMemcpyAsync(outp + recon_n + (size_t)BB * 128 * NN, tokens,
                 (size_t)BB * TCC * LL * sizeof(float),
                 hipMemcpyDeviceToDevice, stream);
}
